// AttentionBiLSTM_Additive_28192165331641
// MI455X (gfx1250) — matrix operand order repaired and checked by reading, not compiled, not run on hardware
//
#include <hip/hip_runtime.h>

typedef __bf16 bf16;
typedef bf16  v16bf __attribute__((ext_vector_type(16)));
typedef float v8f   __attribute__((ext_vector_type(8)));
typedef unsigned int u32x4 __attribute__((ext_vector_type(4)));

union Frag { v16bf v; u32x4 q[2]; };

#define BB   32      // batch
#define SS   512     // seq len
#define EE   512     // embed
#define HH   512     // hidden
#define DD   512
#define VAA  256
#define G4   2048    // 4*H
#define MSB  16384   // S*B

// ---------------- converts / packing ----------------

__global__ void k_f32_to_bf16(const float* __restrict__ src, bf16* __restrict__ dst, int n) {
  int i = blockIdx.x * blockDim.x + threadIdx.x;
  if (i < n) dst[i] = (bf16)src[i];
}

// Pack a (K x N row-major f32) weight into WMMA B-fragment order:
// frag f = nt*(K/32)+kt : 32 lanes x 16 bf16; element i of lane L holds
// W[kt*32 + (L/16)*16 + i][nt*16 + (L%16)]
__global__ void k_pack_b(const float* __restrict__ W, bf16* __restrict__ out, int K, int N) {
  int idx = blockIdx.x * blockDim.x + threadIdx.x;   // (N/16)*(K/32)*512
  int i    = idx & 15;
  int lane = (idx >> 4) & 31;
  int rest = idx >> 9;
  int kch  = K >> 5;
  int kt   = rest % kch;
  int nt   = rest / kch;
  int k = kt * 32 + (lane >> 4) * 16 + i;
  int n = nt * 16 + (lane & 15);
  out[idx] = (bf16)W[(long long)k * N + n];
}

// x_bf[(s*B+b)*E + e] = bf16(emb[words[b,s], e])
__global__ void k_gather(const int* __restrict__ words, const float* __restrict__ emb,
                         bf16* __restrict__ xbf) {
  int t = blockIdx.x * blockDim.x + threadIdx.x;     // 16384*512
  int m = t >> 9, e = t & 511;
  int s = m >> 5, b = m & 31;
  int w = words[b * SS + s];
  xbf[t] = (bf16)emb[(long long)w * EE + e];
}

// outputs_bf[(b*S+s)*1024 + k] from out1/out2 (stored (S,B,H))
__global__ void k_build_out(const float* __restrict__ out1, const float* __restrict__ out2,
                            bf16* __restrict__ obf) {
  int t = blockIdx.x * blockDim.x + threadIdx.x;     // 16384*1024
  int m = t >> 10, k = t & 1023;
  int b = m >> 9, s = m & 511;
  float v = (k < 512) ? out1[((long long)s * BB + b) * HH + k]
                      : out2[((long long)s * BB + b) * HH + (k - 512)];
  obf[t] = (bf16)v;
}

// ---------------- generic bf16 WMMA GEMM ----------------
// C[M,N] (+)= A[M,K] * Bp + bias, where Bp is fragment-packed (k_pack_b).
// block tile 64x128, 8 waves (2x4), wave tile 32x32, K-chunk 32.
// A tile staged to LDS via global_load_async_to_lds_b128 (ASYNCcnt);
// B fragments loaded straight from packed global (L2-resident weights).

__global__ __launch_bounds__(256) void k_gemm_bf16(
    const bf16* __restrict__ A, const bf16* __restrict__ Bp,
    float* __restrict__ C, const float* __restrict__ bias,
    int M, int N, int K, int accFlag) {
  __shared__ bf16 Alds[64][32];
  const int tid  = threadIdx.x;
  const int lane = tid & 31;
  const int wave = tid >> 5;
  const int wm0  = (wave >> 2) * 32;
  const int wn0  = (wave & 3) * 32;
  const int n0   = blockIdx.x * 128;
  const int m0   = blockIdx.y * 64;
  const int kch  = K >> 5;                 // number of 32-wide K chunks
  const int ntg0 = (n0 + wn0) >> 4;        // global n-tile index of this wave

  v8f acc[2][2] = {};

  const int arow = tid >> 2;               // 0..63
  const int aseg = (tid & 3) * 8;          // 0,8,16,24
  bf16* adst = &Alds[arow][aseg];
  const unsigned ldsoff = (unsigned)(unsigned long long)(uintptr_t)adst;
  const bool arow_ok = (m0 + arow) < M;

  for (int k0 = 0, kt = 0; k0 < K; k0 += 32, ++kt) {
    __syncthreads();
    if (arow_ok) {
      unsigned long long ga =
          (unsigned long long)(uintptr_t)(A + (long long)(m0 + arow) * K + k0 + aseg);
      asm volatile("global_load_async_to_lds_b128 %0, %1, off"
                   :: "v"(ldsoff), "v"(ga) : "memory");
    } else {
      u32x4 z = {};
      *(u32x4*)adst = z;
    }
    asm volatile("s_wait_asynccnt 0" ::: "memory");
    __syncthreads();

    const int r = lane & 15, half = lane >> 4;
    Frag af[2], bfr[2];
    #pragma unroll
    for (int mt = 0; mt < 2; ++mt) {
      const bf16* p = &Alds[wm0 + mt * 16 + r][half * 16];
      af[mt].q[0] = *(const u32x4*)p;
      af[mt].q[1] = *(const u32x4*)(p + 8);
    }
    #pragma unroll
    for (int nt = 0; nt < 2; ++nt) {
      const bf16* p = Bp + ((((long long)(ntg0 + nt) * kch + kt) * 32 + lane) << 4);
      bfr[nt].q[0] = *(const u32x4*)p;
      bfr[nt].q[1] = *(const u32x4*)(p + 8);
    }
    #pragma unroll
    for (int mt = 0; mt < 2; ++mt)
      #pragma unroll
      for (int nt = 0; nt < 2; ++nt)
        acc[mt][nt] = __builtin_amdgcn_wmma_f32_16x16x32_bf16(
            false, af[mt].v, false, bfr[nt].v, (short)0, acc[mt][nt], false, false);
  }

  const int r = lane & 15, half = lane >> 4;
  #pragma unroll
  for (int mt = 0; mt < 2; ++mt)
    #pragma unroll
    for (int nt = 0; nt < 2; ++nt)
      #pragma unroll
      for (int v = 0; v < 8; ++v) {
        int m = m0 + wm0 + mt * 16 + half * 8 + v;
        int n = n0 + wn0 + nt * 16 + r;
        if (m < M) {
          float val = acc[mt][nt][v];
          if (bias) val += bias[n];
          long long off = (long long)m * N + n;
          if (accFlag) val += C[off];
          C[off] = val;
        }
      }
}

// ---------------- persistent forward LSTM scan ----------------
// grid=2 (batch halves of 16 rows), block=1024 (32 waves).
// wave w owns hidden columns [w*16, w*16+16) across all four gates;
// h (16x512 bf16) lives in LDS, c lives in registers, Wh_f is fragment-packed.

__global__ __launch_bounds__(1024) void k_fwd_scan(
    const float* __restrict__ pre_in, const bf16* __restrict__ whp,
    float* __restrict__ out1, float* __restrict__ hfin, bf16* __restrict__ hfin_bf) {
  __shared__ bf16 hlds[16][512];
  const int tid  = threadIdx.x;
  const int lane = tid & 31;
  const int wave = tid >> 5;
  const int bg   = blockIdx.x;
  const int r    = lane & 15, half = lane >> 4;
  const int jc   = wave * 16 + r;

  for (int i = tid; i < 16 * 512; i += 1024) ((bf16*)hlds)[i] = (bf16)0.f;
  float cr[8] = {};
  float hreg[8] = {};
  __syncthreads();

  for (int s = 0; s < SS; ++s) {
    v8f acc[4] = {};
    for (int kt = 0; kt < 16; ++kt) {
      Frag af;
      const bf16* pa = &hlds[r][kt * 32 + half * 16];
      af.q[0] = *(const u32x4*)pa;
      af.q[1] = *(const u32x4*)(pa + 8);
      #pragma unroll
      for (int g = 0; g < 4; ++g) {
        int nt = g * 32 + wave;
        Frag bf_;
        const bf16* pb = whp + (((long long)(nt * 16 + kt) * 32 + lane) << 4);
        bf_.q[0] = *(const u32x4*)pb;
        bf_.q[1] = *(const u32x4*)(pb + 8);
        acc[g] = __builtin_amdgcn_wmma_f32_16x16x32_bf16(
            false, af.v, false, bf_.v, (short)0, acc[g], false, false);
      }
    }
    #pragma unroll
    for (int v = 0; v < 8; ++v) {
      int m = half * 8 + v;
      long long base = ((long long)s * BB + bg * 16 + m) * G4;
      float rp = acc[0][v] + pre_in[base + jc];
      float fp = acc[1][v] + pre_in[base + 512 + jc];
      float gp = acc[2][v] + pre_in[base + 1024 + jc];
      float op = acc[3][v] + pre_in[base + 1536 + jc];
      float rg = 1.f / (1.f + __expf(-rp));
      float fg = 1.f / (1.f + __expf(-fp));
      float gg = tanhf(gp);
      float og = 1.f / (1.f + __expf(-op));
      cr[v] = fg * cr[v] + rg * gg;
      float h = og * tanhf(cr[v]);
      hreg[v] = h;
      out1[((long long)s * BB + bg * 16 + m) * HH + jc] = h;
    }
    __syncthreads();
    #pragma unroll
    for (int v = 0; v < 8; ++v) hlds[half * 8 + v][jc] = (bf16)hreg[v];
    __syncthreads();
  }
  #pragma unroll
  for (int v = 0; v < 8; ++v) {
    int m = bg * 16 + half * 8 + v;
    hfin[m * HH + jc]    = hreg[v];
    hfin_bf[m * HH + jc] = (bf16)hreg[v];
  }
}

// ---------------- backward direction: pure elementwise reverse scan ----------------
__global__ void k_bwd_scan(const float* __restrict__ pre_b, float* __restrict__ out2) {
  int t = blockIdx.x * blockDim.x + threadIdx.x;   // B*H = 16384
  int b = t >> 9, j = t & 511;
  float c2 = 0.f;
  for (int s = SS - 1; s >= 0; --s) {
    long long base = ((long long)s * BB + b) * G4;
    float rg = 1.f / (1.f + __expf(-pre_b[base + j]));
    float fg = 1.f / (1.f + __expf(-pre_b[base + 512 + j]));
    float gg = tanhf(pre_b[base + 1024 + j]);
    float og = 1.f / (1.f + __expf(-pre_b[base + 1536 + j]));
    c2 = fg * c2 + rg * gg;
    out2[((long long)s * BB + b) * HH + j] = og * tanhf(c2);
  }
}

// ---------------- small dense / attention tail kernels ----------------

__global__ void k_final_hidden(const float* __restrict__ hfin, const float* __restrict__ W,
                               const float* __restrict__ bias, float* __restrict__ fh) {
  int t = blockIdx.x * blockDim.x + threadIdx.x;   // 32*512
  int b = t >> 9, d = t & 511;
  float acc = bias[d];
  for (int h = 0; h < HH; ++h) acc += hfin[b * HH + h] * W[h * DD + d];
  fh[t] = acc;
}

__global__ void k_wsproj(const float* __restrict__ fh, const float* __restrict__ W,
                         const float* __restrict__ bias, float* __restrict__ ws) {
  int t = blockIdx.x * blockDim.x + threadIdx.x;   // 32*256
  int b = t >> 8, v = t & 255;
  float acc = bias[v];
  for (int d = 0; d < DD; ++d) acc += fh[b * DD + d] * W[d * VAA + v];
  ws[t] = acc;
}

__global__ void k_score(const float* __restrict__ WH, const float* __restrict__ ws,
                        const float* __restrict__ vw, const float* __restrict__ vb,
                        float* __restrict__ score) {
  int t = blockIdx.x * blockDim.x + threadIdx.x;   // B*S = 16384 (b-major)
  int b = t >> 9;
  float acc = 0.f;
  for (int v = 0; v < VAA; ++v)
    acc += tanhf(WH[(long long)t * VAA + v] + ws[b * VAA + v]) * vw[v];
  score[t] = acc + vb[0];
}

__global__ __launch_bounds__(512) void k_softmax(const float* __restrict__ score,
                                                 float* __restrict__ dist) {
  __shared__ float red[512];
  int b = blockIdx.x, s = threadIdx.x;
  float v = score[b * SS + s];
  red[s] = v; __syncthreads();
  for (int o = 256; o > 0; o >>= 1) { if (s < o) red[s] = fmaxf(red[s], red[s + o]); __syncthreads(); }
  float mx = red[0]; __syncthreads();
  float e = __expf(v - mx);
  red[s] = e; __syncthreads();
  for (int o = 256; o > 0; o >>= 1) { if (s < o) red[s] += red[s + o]; __syncthreads(); }
  dist[b * SS + s] = e / red[0];
}

__global__ void k_att(const float* __restrict__ out1, const float* __restrict__ out2,
                      const float* __restrict__ dist, float* __restrict__ att) {
  int t = blockIdx.x * blockDim.x + threadIdx.x;   // 32*1024
  int b = t >> 10, h2 = t & 1023;
  float acc = 0.f;
  for (int s = 0; s < SS; ++s) {
    float x = (h2 < 512) ? out1[((long long)s * BB + b) * HH + h2]
                         : out2[((long long)s * BB + b) * HH + (h2 - 512)];
    acc += x * dist[b * SS + s];
  }
  att[t] = acc;
}

__global__ void k_headA(const float* __restrict__ fh, const float* __restrict__ att,
                        const float* __restrict__ W, const float* __restrict__ bias,
                        float* __restrict__ hid) {
  int t = blockIdx.x * blockDim.x + threadIdx.x;   // 32*768
  int b = t / 768, u = t % 768;
  float acc = bias[u];
  for (int k = 0; k < 1536; ++k) {
    float x = (k < 512) ? fh[b * DD + k] : att[b * 1024 + (k - 512)];
    acc += x * W[k * 768 + u];
  }
  hid[t] = acc;
}

__global__ void k_headB(const float* __restrict__ hid, const float* __restrict__ W,
                        const float* __restrict__ bias, float* __restrict__ out) {
  int t = blockIdx.x * blockDim.x + threadIdx.x;   // 32*2
  if (t >= 64) return;
  int b = t >> 1, o = t & 1;
  float acc = bias[o];
  for (int u = 0; u < 768; ++u) acc += hid[b * 768 + u] * W[u * 2 + o];
  out[t] = 1.f / (1.f + __expf(-acc));
}

// ---------------- launch ----------------

extern "C" void kernel_launch(void* const* d_in, const int* in_sizes, int n_in,
                              void* d_out, int out_size, void* d_ws, size_t ws_size,
                              hipStream_t stream) {
  (void)in_sizes; (void)n_in; (void)out_size; (void)ws_size;
  const int*   words  = (const int*)  d_in[0];
  const float* emb    = (const float*)d_in[1];
  const float* Wi_f   = (const float*)d_in[2];
  const float* Wh_f   = (const float*)d_in[3];
  const float* b_f    = (const float*)d_in[4];
  const float* Wi_b   = (const float*)d_in[5];
  const float* Wh_b   = (const float*)d_in[6];
  const float* b_b    = (const float*)d_in[7];
  const float* W_lo   = (const float*)d_in[8];
  const float* b_lo   = (const float*)d_in[9];
  const float* W_ah   = (const float*)d_in[10];
  const float* b_ah   = (const float*)d_in[11];
  const float* W_as   = (const float*)d_in[12];
  const float* b_as   = (const float*)d_in[13];
  const float* v_w    = (const float*)d_in[14];
  const float* v_b    = (const float*)d_in[15];
  const float* W_A    = (const float*)d_in[16];
  const float* b_A    = (const float*)d_in[17];
  const float* W_B    = (const float*)d_in[18];
  const float* b_B    = (const float*)d_in[19];

  float* sig_out = (float*)d_out;          // (32,2)
  float* hfin    = (float*)d_out + 64;     // (32,512) f32, tuple element 2

  char* base = (char*)d_ws;
  size_t off = 0;
  auto alloc = [&](size_t bytes) -> char* {
    off = (off + 255) & ~(size_t)255;
    char* p = base + off; off += bytes; return p;
  };
  bf16*  xbf     = (bf16*) alloc((size_t)MSB * EE * 2);
  bf16*  WiFp    = (bf16*) alloc((size_t)EE * G4 * 2);   // fragment-packed
  bf16*  WiBp    = (bf16*) alloc((size_t)EE * G4 * 2);
  bf16*  WhBp    = (bf16*) alloc((size_t)HH * G4 * 2);
  bf16*  Wahp    = (bf16*) alloc((size_t)1024 * VAA * 2);
  bf16*  whp     = (bf16*) alloc((size_t)HH * G4 * 2);
  float* pre_in  = (float*)alloc((size_t)MSB * G4 * 4);
  float* pre_b   = (float*)alloc((size_t)MSB * G4 * 4);
  float* out1    = (float*)alloc((size_t)MSB * HH * 4);
  float* out2    = (float*)alloc((size_t)MSB * HH * 4);
  bf16*  obf     = (bf16*) alloc((size_t)MSB * 1024 * 2);
  bf16*  hfin_bf = (bf16*) alloc((size_t)BB * HH * 2);
  float* WHb     = (float*)alloc((size_t)MSB * VAA * 4);
  float* WSb     = (float*)alloc((size_t)BB * VAA * 4);
  float* fhid    = (float*)alloc((size_t)BB * DD * 4);
  float* score   = (float*)alloc((size_t)BB * SS * 4);
  float* dist    = (float*)alloc((size_t)BB * SS * 4);
  float* att     = (float*)alloc((size_t)BB * 1024 * 4);
  float* hidA    = (float*)alloc((size_t)BB * 768 * 4);

  // weight packing into WMMA fragment order
  k_pack_b<<<4096, 256, 0, stream>>>(Wi_f, WiFp, EE, G4);
  k_pack_b<<<4096, 256, 0, stream>>>(Wi_b, WiBp, EE, G4);
  k_pack_b<<<4096, 256, 0, stream>>>(Wh_b, WhBp, HH, G4);
  k_pack_b<<<1024, 256, 0, stream>>>(W_ah, Wahp, 1024, VAA);
  k_pack_b<<<4096, 256, 0, stream>>>(Wh_f, whp, HH, G4);
  k_gather<<<32768, 256, 0, stream>>>(words, emb, xbf);

  // pre_in = x @ Wi_f + b_f ;  pre_b = x @ Wi_b + b_b
  k_gemm_bf16<<<dim3(G4 / 128, MSB / 64), 256, 0, stream>>>(xbf, WiFp, pre_in, b_f, MSB, G4, EE, 0);
  k_gemm_bf16<<<dim3(G4 / 128, MSB / 64), 256, 0, stream>>>(xbf, WiBp, pre_b,  b_b, MSB, G4, EE, 0);

  // forward recurrence (also writes h_final into d_out)
  k_fwd_scan<<<2, 1024, 0, stream>>>(pre_in, whp, out1, hfin, hfin_bf);

  // pre_b += h_final @ Wh_b ; then elementwise reverse scan
  k_gemm_bf16<<<dim3(G4 / 128, 1), 256, 0, stream>>>(hfin_bf, WhBp, pre_b, nullptr, BB, G4, HH, 1);
  k_bwd_scan<<<64, 256, 0, stream>>>(pre_b, out2);

  // attention
  k_build_out<<<65536, 256, 0, stream>>>(out1, out2, obf);
  k_gemm_bf16<<<dim3(VAA / 128, MSB / 64), 256, 0, stream>>>(obf, Wahp, WHb, b_ah, MSB, VAA, 1024, 0);
  k_final_hidden<<<64, 256, 0, stream>>>(hfin, W_lo, b_lo, fhid);
  k_wsproj<<<32, 256, 0, stream>>>(fhid, W_as, b_as, WSb);
  k_score<<<64, 256, 0, stream>>>(WHb, WSb, v_w, v_b, score);
  k_softmax<<<32, 512, 0, stream>>>(score, dist);
  k_att<<<128, 256, 0, stream>>>(out1, out2, dist, att);

  // output head
  k_headA<<<96, 256, 0, stream>>>(fhid, att, W_A, b_A, hidA);
  k_headB<<<1, 64, 0, stream>>>(hidA, W_B, b_B, sig_out);
}
